// build_model_17523466567932
// MI455X (gfx1250) — matrix operand order repaired and checked by reading, not compiled, not run on hardware
//
#include <hip/hip_runtime.h>
#include <hip/hip_bf16.h>

typedef __attribute__((ext_vector_type(16))) _Float16 v16h;
typedef __attribute__((ext_vector_type(8)))  _Float16 v8h;
typedef __attribute__((ext_vector_type(8)))  float    v8f;

#define B_  1024
#define T_  128
#define I_  128
#define H_  256
#define OUT_ 64

__device__ __forceinline__ float sigmoidf_(float x) {
    return 1.0f / (1.0f + __expf(-x));
}

__device__ __forceinline__ v8f wmma_f16(v16h a, v16h b, v8f c) {
    // D = A(16x32 f16) * B(32x16 f16) + C(16x16 f32)
    return __builtin_amdgcn_wmma_f32_16x16x32_f16(
        /*neg_a=*/false, a, /*neg_b=*/false, b,
        /*c_mod=*/(short)0, c, /*reuse_a=*/false, /*reuse_b=*/false);
}

// A fragment (16x32 MxK) from f32 row (fallback path): converts on the fly.
__device__ __forceinline__ v16h load_a_f32(const float* __restrict__ row, int kc, int lane) {
    const int kb = (lane & 16) ? 16 : 0;
    v16h a;
#pragma unroll
    for (int i = 0; i < 8; ++i) a[i]     = (_Float16)row[kc + kb + i];
#pragma unroll
    for (int i = 0; i < 8; ++i) a[8 + i] = (_Float16)row[kc + kb + 8 + i];
    return a;
}

// A fragment (16x32 MxK) from f16 row: two 16-byte vector loads, no cvts.
__device__ __forceinline__ v16h load_a_f16(const _Float16* __restrict__ row, int kc, int lane) {
    const int kb = (lane & 16) ? 16 : 0;
    const v8h lo = *(const v8h*)(row + kc + kb);
    const v8h hi = *(const v8h*)(row + kc + kb + 8);
    v16h a;
#pragma unroll
    for (int i = 0; i < 8; ++i) { a[i] = lo[i]; a[8 + i] = hi[i]; }
    return a;
}

// B fragment (32x16 KxN), B = W^T, W row-major [rows x K] f16: 16 contiguous halves.
__device__ __forceinline__ v16h load_b_f16(const _Float16* __restrict__ w, int j0, int K,
                                           int kc, int lane) {
    const int n  = lane & 15;
    const int kb = (lane & 16) ? 16 : 0;
    return *(const v16h*)(w + (size_t)(j0 + n) * K + kc + kb);
}

// ---- one-time per-call staging kernels ----

__global__ void cvt_weights(const float* __restrict__ wih, const float* __restrict__ whh,
                            const float* __restrict__ wst,
                            _Float16* __restrict__ wih_h, _Float16* __restrict__ whh_h,
                            _Float16* __restrict__ wst_h) {
    const int n1 = 3 * H_ * I_;
    const int n2 = 3 * H_ * H_;
    const int n3 = H_ * H_;
    const int total = n1 + n2 + n3;
    for (int idx = blockIdx.x * blockDim.x + threadIdx.x; idx < total;
         idx += gridDim.x * blockDim.x) {
        if (idx < n1)            wih_h[idx]           = (_Float16)wih[idx];
        else if (idx < n1 + n2)  whh_h[idx - n1]      = (_Float16)whh[idx - n1];
        else                     wst_h[idx - n1 - n2] = (_Float16)wst[idx - n1 - n2];
    }
}

__global__ void cvt_f32_to_f16(const float* __restrict__ src, _Float16* __restrict__ dst,
                               size_t n) {
    for (size_t idx = (size_t)blockIdx.x * blockDim.x + threadIdx.x; idx < n;
         idx += (size_t)gridDim.x * blockDim.x) {
        dst[idx] = (_Float16)src[idx];
    }
}

// ---- per-step kernels ----

// Kernel 1: h_new = GRU(x_t, h).  One wave = one 16x16 tile of [B x H].
// Four f32 accumulators: r,z combine gi+gh; i_n (x path) and h_n (h path) kept
// separate because r gates only the Whh*h contribution to n.
template <bool XF16>
__global__ void __launch_bounds__(256)
gru_gates_kernel(const float* __restrict__ h_in,       // [B,H] f32 (s at t==0)
                 const _Float16* __restrict__ h16_in,  // [B,H] f16 mirror
                 const float* __restrict__ x32,        // [B,T,I] f32 (fallback)
                 const _Float16* __restrict__ x16,     // [B,T,I] f16 (fast path)
                 const _Float16* __restrict__ wih,     // [3H,I] f16
                 const _Float16* __restrict__ whh,     // [3H,H] f16
                 float* __restrict__ h_new,            // [B,H] f32
                 _Float16* __restrict__ h16_new,       // [B,H] f16 mirror
                 int t) {
    const int wave = (blockIdx.x * blockDim.x + threadIdx.x) >> 5;
    const int lane = threadIdx.x & 31;
    const int jt = wave & (H_ / 16 - 1);   // weight-column tiles fastest (L2 reuse)
    const int bt = wave >> 4;
    const int b0 = bt * 16, j0 = jt * 16;
    const int m  = lane & 15;

    const size_t xoff = ((size_t)(b0 + m) * T_ + t) * I_;
    const size_t hoff = (size_t)(b0 + m) * H_;

    v8f acc_r = {}, acc_z = {}, acc_in = {}, acc_hn = {};

    // gi = x_t @ w_ih^T   (K = I = 128 -> 4 chunks)
#pragma unroll
    for (int kc = 0; kc < I_; kc += 32) {
        v16h a = XF16 ? load_a_f16(x16 + xoff, kc, lane)
                      : load_a_f32(x32 + xoff, kc, lane);
        v16h br = load_b_f16(wih, j0,          I_, kc, lane);
        v16h bz = load_b_f16(wih, j0 + H_,     I_, kc, lane);
        v16h bn = load_b_f16(wih, j0 + 2 * H_, I_, kc, lane);
        acc_r  = wmma_f16(a, br, acc_r);
        acc_z  = wmma_f16(a, bz, acc_z);
        acc_in = wmma_f16(a, bn, acc_in);
    }
    // gh = h @ w_hh^T     (K = H = 256 -> 8 chunks), A from f16 mirror
#pragma unroll
    for (int kc = 0; kc < H_; kc += 32) {
        v16h a  = load_a_f16(h16_in + hoff, kc, lane);
        v16h br = load_b_f16(whh, j0,          H_, kc, lane);
        v16h bz = load_b_f16(whh, j0 + H_,     H_, kc, lane);
        v16h bn = load_b_f16(whh, j0 + 2 * H_, H_, kc, lane);
        acc_r  = wmma_f16(a, br, acc_r);
        acc_z  = wmma_f16(a, bz, acc_z);
        acc_hn = wmma_f16(a, bn, acc_hn);
    }

    // C/D layout: reg v -> row v (+8 for hi lanes), col = lane%16
    const int n    = lane & 15;
    const int mrow = (lane >> 4) * 8;
#pragma unroll
    for (int v = 0; v < 8; ++v) {
        const size_t idx = (size_t)(b0 + mrow + v) * H_ + j0 + n;
        const float hold = h_in[idx];
        const float r  = sigmoidf_(acc_r[v]);
        const float z  = sigmoidf_(acc_z[v]);
        const float nn = tanhf(acc_in[v] + r * acc_hn[v]);
        const float hv = (1.0f - z) * nn + z * hold;
        h_new[idx]   = hv;
        h16_new[idx] = (_Float16)hv;
    }
}

// Kernel 2: s_next = sigmoid(h_new @ w_state^T) -> output slice + next-step carry.
// Wave 0 of block 0 additionally computes the reward head for batch row 0.
__global__ void __launch_bounds__(256)
state_kernel(const float* __restrict__ h_new,        // [B,H] f32 (reward head)
             const _Float16* __restrict__ h16_new,   // [B,H] f16 (WMMA A)
             const _Float16* __restrict__ wst,       // [H,H] f16
             const float* __restrict__ w_reward,     // [OUT,H] f32
             float* __restrict__ h_cur,              // [B,H] f32 carry
             _Float16* __restrict__ h16_cur,         // [B,H] f16 carry mirror
             float* __restrict__ s_out,              // [B,T,H] flat
             float* __restrict__ r_out,              // [T,OUT] flat
             int t) {
    const int wave = (blockIdx.x * blockDim.x + threadIdx.x) >> 5;
    const int lane = threadIdx.x & 31;
    const int jt = wave & (H_ / 16 - 1);
    const int bt = wave >> 4;
    const int b0 = bt * 16, j0 = jt * 16;
    const int m  = lane & 15;

    const _Float16* hnrow = h16_new + (size_t)(b0 + m) * H_;

    v8f acc = {};
#pragma unroll
    for (int kc = 0; kc < H_; kc += 32) {
        v16h a = load_a_f16(hnrow, kc, lane);
        v16h b = load_b_f16(wst, j0, H_, kc, lane);
        acc = wmma_f16(a, b, acc);
    }

    const int n    = lane & 15;
    const int mrow = (lane >> 4) * 8;
#pragma unroll
    for (int v = 0; v < 8; ++v) {
        const int   bi = b0 + mrow + v;
        const float sv = sigmoidf_(acc[v]);
        h_cur[(size_t)bi * H_ + j0 + n]   = sv;
        h16_cur[(size_t)bi * H_ + j0 + n] = (_Float16)sv;
        s_out[((size_t)bi * T_ + t) * H_ + j0 + n] = sv;
    }

    // reward head: r_t = sigmoid(h_new[0] @ w_reward^T), OUT=64
    if (blockIdx.x == 0 && (threadIdx.x >> 5) == 0) {
#pragma unroll
        for (int o = lane; o < OUT_; o += 32) {
            float a = 0.0f;
            const float* wr = w_reward + (size_t)o * H_;
            for (int k = 0; k < H_; ++k) a += h_new[k] * wr[k];
            r_out[(size_t)t * OUT_ + o] = sigmoidf_(a);
        }
    }
}

extern "C" void kernel_launch(void* const* d_in, const int* in_sizes, int n_in,
                              void* d_out, int out_size, void* d_ws, size_t ws_size,
                              hipStream_t stream) {
    (void)in_sizes; (void)n_in; (void)out_size;
    const float* s        = (const float*)d_in[0];   // [B,H]
    const float* a_list   = (const float*)d_in[1];   // [B,T,I]
    const float* w_ih     = (const float*)d_in[2];   // [3H,I]
    const float* w_hh     = (const float*)d_in[3];   // [3H,H]
    const float* w_reward = (const float*)d_in[4];   // [OUT,H]
    const float* w_state  = (const float*)d_in[5];   // [H,H]

    float* r_out = (float*)d_out;                    // [1,T,OUT] flat
    float* s_out = r_out + (size_t)T_ * OUT_;        // [1,B,T,H] flat

    // workspace layout:
    //   h_cur f32 | h_new f32 | h16_cur | h16_new | f16 weights | (optional) x16
    const size_t BH = (size_t)B_ * H_;
    float*    h_cur   = (float*)d_ws;
    float*    h_new   = h_cur + BH;
    _Float16* h16_cur = (_Float16*)(h_new + BH);
    _Float16* h16_new = h16_cur + BH;
    _Float16* wih_h   = h16_new + BH;
    _Float16* whh_h   = wih_h + (size_t)3 * H_ * I_;
    _Float16* wst_h   = whh_h + (size_t)3 * H_ * H_;
    _Float16* x16     = wst_h + (size_t)H_ * H_;

    const size_t base_bytes = (size_t)((char*)x16 - (char*)d_ws);
    const size_t x16_bytes  = (size_t)B_ * T_ * I_ * sizeof(_Float16);
    const bool   use_x16    = (ws_size >= base_bytes + x16_bytes);

    cvt_weights<<<512, 256, 0, stream>>>(w_ih, w_hh, w_state, wih_h, whh_h, wst_h);
    // f16 mirror of the initial carry s (A operand of the first gh GEMM)
    cvt_f32_to_f16<<<512, 256, 0, stream>>>(s, h16_cur, BH);
    if (use_x16) {
        cvt_f32_to_f16<<<2048, 256, 0, stream>>>(a_list, x16, (size_t)B_ * T_ * I_);
    }

    const dim3 grid(128), block(256);                // 1024 waves = 64 x 16 tiles
    for (int t = 0; t < T_; ++t) {
        const float* h_in = (t == 0) ? s : h_cur;
        if (use_x16) {
            gru_gates_kernel<true><<<grid, block, 0, stream>>>(
                h_in, h16_cur, a_list, x16, wih_h, whh_h, h_new, h16_new, t);
        } else {
            gru_gates_kernel<false><<<grid, block, 0, stream>>>(
                h_in, h16_cur, a_list, x16, wih_h, whh_h, h_new, h16_new, t);
        }
        state_kernel<<<grid, block, 0, stream>>>(
            h_new, h16_new, wst_h, w_reward, h_cur, h16_cur, s_out, r_out, t);
    }
}
